// BitLinear_4767413698790
// MI455X (gfx1250) — compile-verified, matrix operand order checked by reading, not run on hardware
//
#include <hip/hip_runtime.h>
#include <hip/hip_bf16.h>

// ---------------------------------------------------------------------------
// BitLinear on MI455X (gfx1250, wave32):
//   scale = max(mean|W|, 1e-6); Wq = sign(W/scale)*(|W/scale|>0.5)*scale
//   out   = x @ Wq^T + bias
// Bandwidth-bound (~0.9 GB HBM traffic -> ~38us at 23.3 TB/s). Pre-pass
// converts x->bf16 (64MB) and W->ternary*scale bf16 (128MB); both fit the
// 192MB L2 together. GEMM uses v_wmma_f32_16x16x32_bf16 with double-buffered
// LDS tiles fed by GLOBAL_LOAD_ASYNC_TO_LDS_B128 (no VGPR staging, no spills).
// ---------------------------------------------------------------------------

#define M_DIM 8192
#define N_DIM 16384
#define K_DIM 4096

#define TILE_M 128
#define TILE_N 128
#define TILE_K 32
#define LDT    40              // padded LDS row stride (ushorts): 32 + 8

typedef __attribute__((ext_vector_type(16))) __bf16 v16bf;
typedef __attribute__((ext_vector_type(8)))  __bf16 v8bf;
typedef __attribute__((ext_vector_type(8)))  float  v8f;
typedef __attribute__((ext_vector_type(4)))  int    v4i;

typedef __attribute__((address_space(1))) v4i* g_v4i_ptr;   // global
typedef __attribute__((address_space(3))) v4i* l_v4i_ptr;   // LDS

// ---------------- CDNA5 async global->LDS helpers --------------------------
__device__ __forceinline__ void async_copy_b128(const void* gsrc, void* lds_dst) {
#if __has_builtin(__builtin_amdgcn_global_load_async_to_lds_b128)
    __builtin_amdgcn_global_load_async_to_lds_b128(
        (g_v4i_ptr)(void*)gsrc,          // global v4i* (AS1), const dropped
        (l_v4i_ptr)lds_dst,              // LDS    v4i* (AS3)
        /*offset=*/0, /*cpol=*/0);
#else
    unsigned lds_off =
        (unsigned)(__SIZE_TYPE__)(__attribute__((address_space(3))) char*)lds_dst;
    asm volatile("global_load_async_to_lds_b128 %0, %1, off"
                 :
                 : "v"(lds_off), "v"(gsrc)
                 : "memory");
#endif
}

__device__ __forceinline__ void wait_async0() {
#if __has_builtin(__builtin_amdgcn_s_wait_asynccnt)
    __builtin_amdgcn_s_wait_asynccnt(0);
#else
    asm volatile("s_wait_asynccnt 0" ::: "memory");
#endif
}

// ---------------- bf16 pack helpers ----------------
__device__ __forceinline__ unsigned short f2bf_bits(float f) {
    __bf16 h = (__bf16)f;                 // RNE convert
    unsigned short u;
    __builtin_memcpy(&u, &h, sizeof(u));
    return u;
}
__device__ __forceinline__ uint2 pack4_bf16(float a, float b, float c, float d) {
    uint2 r;
    r.x = (unsigned)f2bf_bits(a) | ((unsigned)f2bf_bits(b) << 16);
    r.y = (unsigned)f2bf_bits(c) | ((unsigned)f2bf_bits(d) << 16);
    return r;
}

// ---------------- 1) per-block partial sums of |W| (deterministic) ----------
__global__ __launch_bounds__(256) void k_abs_partial(const float* __restrict__ w,
                                                     float* __restrict__ partials,
                                                     int n) {
    __shared__ float sm[256];
    float acc = 0.0f;
    const int stride = gridDim.x * 256;
    for (int i = blockIdx.x * 256 + threadIdx.x; i < n; i += stride)
        acc += fabsf(w[i]);
    sm[threadIdx.x] = acc;
    __syncthreads();
    #pragma unroll
    for (int s = 128; s > 0; s >>= 1) {
        if (threadIdx.x < s) sm[threadIdx.x] += sm[threadIdx.x + s];
        __syncthreads();
    }
    if (threadIdx.x == 0) partials[blockIdx.x] = sm[0];
}

// ---------------- 2) finalize scale ----------------------------------------
__global__ __launch_bounds__(256) void k_scale_finalize(const float* __restrict__ partials,
                                                        float* __restrict__ scale,
                                                        int nparts, float invN) {
    __shared__ float sm[256];
    float acc = 0.0f;
    for (int i = threadIdx.x; i < nparts; i += 256) acc += partials[i];
    sm[threadIdx.x] = acc;
    __syncthreads();
    #pragma unroll
    for (int s = 128; s > 0; s >>= 1) {
        if (threadIdx.x < s) sm[threadIdx.x] += sm[threadIdx.x + s];
        __syncthreads();
    }
    if (threadIdx.x == 0) scale[0] = fmaxf(sm[0] * invN, 1e-6f);
}

// ---------------- 3) x (f32) -> bf16 ---------------------------------------
__global__ __launch_bounds__(256) void k_conv_x(const float4* __restrict__ x,
                                                uint2* __restrict__ xq, int n4) {
    const int stride = gridDim.x * blockDim.x;
    for (int i = blockIdx.x * blockDim.x + threadIdx.x; i < n4; i += stride) {
        float4 v = x[i];
        xq[i] = pack4_bf16(v.x, v.y, v.z, v.w);
    }
}

// ---------------- 4) W -> ternary*scale (bf16) -----------------------------
__global__ __launch_bounds__(256) void k_quant_w(const float4* __restrict__ w,
                                                 uint2* __restrict__ wq,
                                                 const float* __restrict__ scale_p,
                                                 int n4) {
    const float s   = scale_p[0];
    const float inv = 1.0f / s;
    const int stride = gridDim.x * blockDim.x;
    for (int i = blockIdx.x * blockDim.x + threadIdx.x; i < n4; i += stride) {
        float4 v = w[i];
        float qa = (fabsf(v.x * inv) > 0.5f) ? copysignf(s, v.x) : 0.0f;
        float qb = (fabsf(v.y * inv) > 0.5f) ? copysignf(s, v.y) : 0.0f;
        float qc = (fabsf(v.z * inv) > 0.5f) ? copysignf(s, v.z) : 0.0f;
        float qd = (fabsf(v.w * inv) > 0.5f) ? copysignf(s, v.w) : 0.0f;
        wq[i] = pack4_bf16(qa, qb, qc, qd);
    }
}

// ---------------- fragment load (16-bit A/B 16x32 ISA layout) --------------
// lanes 0-15 : row = lane,    K 0..7 (chunk0) and 16..23 (chunk1)
// lanes 16-31: row = lane-16, K 8..15 and 24..31 -> base offset +8 elements
__device__ __forceinline__ v16bf load_frag(const unsigned short* p) {
    v8bf c0 = *(const v8bf*)(p);        // ds_load_b128
    v8bf c1 = *(const v8bf*)(p + 16);   // ds_load_b128
    return __builtin_shufflevector(c0, c1, 0, 1, 2, 3, 4, 5, 6, 7,
                                           8, 9, 10, 11, 12, 13, 14, 15);
}

// ---------------- 5) GEMM: out = xq @ wq^T + bias --------------------------
__global__ __launch_bounds__(256) void k_gemm_bf16(const unsigned short* __restrict__ xq,
                                                   const unsigned short* __restrict__ wq,
                                                   const float* __restrict__ bias,
                                                   float* __restrict__ out) {
    __shared__ unsigned short As[2][TILE_M * LDT];
    __shared__ unsigned short Bs[2][TILE_M * LDT];

    const int tid   = threadIdx.x;
    const int lane  = tid & 31;
    const int wid   = tid >> 5;          // 0..7
    const int lrow  = lane & 15;
    const int lhalf = (lane >> 4) & 1;

    const int m0 = blockIdx.y * TILE_M;
    const int n0 = blockIdx.x * TILE_N;

    const int wave_m = (wid & 1) * 64;   // 2 waves along M (64 rows each)
    const int wave_n = (wid >> 1) * 32;  // 4 waves along N (32 cols each)

    v8f acc[4][2] = {};

    // Issue async copies for one K-block: 128 rows x 32 bf16 = 512 x 16B
    // chunks per matrix, 2 chunks per thread per matrix. No VGPR staging.
    auto async_tile = [&](int kb, int buf) {
        const int kbase = kb * TILE_K;
        #pragma unroll
        for (int i = 0; i < 2; ++i) {
            const int c   = tid + i * 256;
            const int row = c >> 2;
            const int kc  = (c & 3) * 8;
            async_copy_b128(xq + (size_t)(m0 + row) * K_DIM + kbase + kc,
                            &As[buf][row * LDT + kc]);
            async_copy_b128(wq + (size_t)(n0 + row) * K_DIM + kbase + kc,
                            &Bs[buf][row * LDT + kc]);
        }
    };

    // prologue: stage K-block 0
    async_tile(0, 0);
    wait_async0();
    __syncthreads();

    const int NK = K_DIM / TILE_K;       // 128 K-steps
    for (int kb = 0; kb < NK; ++kb) {
        const int cur = kb & 1;
        // prefetch next K-block into the other buffer (safe: last read of
        // that buffer finished before the previous barrier)
        if (kb + 1 < NK) async_tile(kb + 1, cur ^ 1);

        // B fragments (2 per wave), reused across 4 M sub-tiles
        v16bf bfrag[2];
        #pragma unroll
        for (int ni = 0; ni < 2; ++ni)
            bfrag[ni] = load_frag(&Bs[cur][(wave_n + ni * 16 + lrow) * LDT + lhalf * 8]);

        #pragma unroll
        for (int mi = 0; mi < 4; ++mi) {
            v16bf afrag = load_frag(&As[cur][(wave_m + mi * 16 + lrow) * LDT + lhalf * 8]);
            #pragma unroll
            for (int ni = 0; ni < 2; ++ni)
                acc[mi][ni] = __builtin_amdgcn_wmma_f32_16x16x32_bf16(
                    /*neg_a=*/false, afrag, /*neg_b=*/false, bfrag[ni],
                    /*c_mod=*/(short)0, acc[mi][ni],
                    /*reuse_a=*/false, /*reuse_b=*/false);
        }

        wait_async0();        // async writes to next buffer landed in LDS
        __syncthreads();      // all waves done reading cur / writing nxt
    }

    // epilogue: C/D layout -> VGPR r holds (M = r + 8*lhalf, N = lrow); + bias
    const int mg = m0 + wave_m;
    const int ng = n0 + wave_n;
    #pragma unroll
    for (int ni = 0; ni < 2; ++ni) {
        const int N  = ng + ni * 16 + lrow;
        const float bv = bias[N];
        #pragma unroll
        for (int mi = 0; mi < 4; ++mi) {
            #pragma unroll
            for (int r = 0; r < 8; ++r) {
                const int M = mg + mi * 16 + lhalf * 8 + r;
                out[(size_t)M * N_DIM + N] = acc[mi][ni][r] + bv;
            }
        }
    }
}

// ---------------------------------------------------------------------------
extern "C" void kernel_launch(void* const* d_in, const int* in_sizes, int n_in,
                              void* d_out, int out_size, void* d_ws, size_t ws_size,
                              hipStream_t stream) {
    const float* x    = (const float*)d_in[0];   // 8192 x 4096
    const float* w    = (const float*)d_in[1];   // 16384 x 4096
    const float* bias = (const float*)d_in[2];   // 16384
    float* out        = (float*)d_out;           // 8192 x 16384

    char* ws = (char*)d_ws;
    // workspace layout (all 16B aligned):
    //   [0,      8KB)  : 2048 reduction partials (f32)
    //   [8KB,   8KB+4) : scale (f32)
    //   [16KB,  +64MB) : x in bf16
    //   [...,  +128MB) : Wq in bf16
    float*          partials = (float*)ws;
    float*          scale    = (float*)(ws + 2048 * sizeof(float));
    unsigned short* xq       = (unsigned short*)(ws + 16384);
    unsigned short* wqp      = (unsigned short*)(ws + 16384 + (size_t)M_DIM * K_DIM * 2);

    const int nW  = N_DIM * K_DIM;       // 67108864
    const int nW4 = nW / 4;
    const int nX4 = (M_DIM * K_DIM) / 4; // 8388608

    // 1) sum |W| partials (fixed-order, deterministic)
    k_abs_partial<<<2048, 256, 0, stream>>>(w, partials, nW);
    // 2) scale = max(mean|W|, 1e-6)
    k_scale_finalize<<<1, 256, 0, stream>>>(partials, scale, 2048, 1.0f / (float)nW);
    // 3) x -> bf16
    k_conv_x<<<4096, 256, 0, stream>>>((const float4*)x, (uint2*)xq, nX4);
    // 4) W -> ternary*scale in bf16
    k_quant_w<<<8192, 256, 0, stream>>>((const float4*)w, (uint2*)wqp, scale, nW4);
    // 5) WMMA GEMM + bias (async-LDS double-buffered)
    dim3 grid(N_DIM / TILE_N, M_DIM / TILE_M);   // 128 x 64 blocks
    k_gemm_bf16<<<grid, 256, 0, stream>>>(xq, wqp, bias, out);
}
